// EncoderDecoderRecallmodel_39737037422865
// MI455X (gfx1250) — compile-verified, matrix operand order checked by reading, not run on hardware
//
#include <hip/hip_runtime.h>

// ---------------------------------------------------------------------------
// Seq2seq LSTM (B=128,T=128,V=1024,E=512,H=1024) for MI455X / gfx1250.
// Per-timestep WMMA cell kernels (bf16 inputs, fp32 accumulate), fused gate
// activations in registers, ping-pong h/c state, decoder argmax feedback.
// ---------------------------------------------------------------------------

typedef __attribute__((ext_vector_type(16))) __bf16 v16bf;
typedef __attribute__((ext_vector_type(8)))  __bf16 v8bf;
typedef __attribute__((ext_vector_type(8)))  float  v8f;

#define B_ 128
#define T_ 128
#define V_ 1024
#define E_ 512
#define H_ 1024

__device__ __forceinline__ float sigmoidf_(float x) {
    return 1.0f / (1.0f + __expf(-x));
}

// Load one 16x32 bf16 WMMA operand fragment (A layout; B is symmetric with
// lane->N).  Per lane: two contiguous 16-byte chunks of row `row`:
//   K = k0 + (lane>=16 ? 8 : 0) + {0..7}  and  +16 + {0..7}
__device__ __forceinline__ v16bf load_frag(const __bf16* __restrict__ base,
                                           int row, int ld, int k0, int lane) {
    int kb = k0 + ((lane >> 4) << 3);
    const __bf16* p = base + (size_t)row * ld + kb;
    v8bf lo = *reinterpret_cast<const v8bf*>(p);
    v8bf hi = *reinterpret_cast<const v8bf*>(p + 16);
    v16bf r;
#pragma unroll
    for (int i = 0; i < 8; ++i) { r[i] = lo[i]; r[i + 8] = hi[i]; }
    return r;
}

// ---------------------------------------------------------------------------
// One LSTM cell step:  gates = x@W_ih^T + h@W_hh^T + b ; fused activations.
// 512 waves: mt in [0,8) x ht in [0,64).  Each wave computes a 16x16 tile of
// all four gates (4 WMMA accumulators) and the resulting h/c tile.
// ---------------------------------------------------------------------------
__global__ __launch_bounds__(128)
void lstm_cell_wmma(const __bf16* __restrict__ x,       // B x E_  (bf16)
                    const __bf16* __restrict__ h_in,    // B x H_  (bf16)
                    const float*  __restrict__ c_in,    // B x H_
                    const __bf16* __restrict__ W_ih,    // 4H x E_ (bf16)
                    const __bf16* __restrict__ W_hh,    // 4H x H_ (bf16)
                    const float*  __restrict__ bias,    // 4H
                    float*  __restrict__ c_out,         // B x H_
                    float*  __restrict__ h_out_f32,     // B x H_
                    __bf16* __restrict__ h_out_bf16)    // B x H_
{
    int tid  = blockIdx.x * blockDim.x + threadIdx.x;
    int lane = tid & 31;
    int wid  = tid >> 5;         // 0..511
    int mt   = wid >> 6;         // 0..7   (M tile: 16 batch rows)
    int ht   = wid & 63;         // 0..63  (N tile: 16 hidden cols)

    v8f acc0 = {}, acc1 = {}, acc2 = {}, acc3 = {};

    int arow = mt * 16 + (lane & 15);   // batch row this lane feeds (A)
    int bcol = ht * 16 + (lane & 15);   // hidden col this lane feeds (B)

    // ---- x @ W_ih^T ------------------------------------------------------
    for (int k0 = 0; k0 < E_; k0 += 32) {
        __builtin_prefetch(x + (size_t)arow * E_ + k0 + 128, 0, 1);
        v16bf a  = load_frag(x,    arow,            E_, k0, lane);
        v16bf b0 = load_frag(W_ih, 0 * H_ + bcol,   E_, k0, lane);
        v16bf b1 = load_frag(W_ih, 1 * H_ + bcol,   E_, k0, lane);
        v16bf b2 = load_frag(W_ih, 2 * H_ + bcol,   E_, k0, lane);
        v16bf b3 = load_frag(W_ih, 3 * H_ + bcol,   E_, k0, lane);
        acc0 = __builtin_amdgcn_wmma_f32_16x16x32_bf16(false, a, false, b0, (short)0, acc0, false, false);
        acc1 = __builtin_amdgcn_wmma_f32_16x16x32_bf16(false, a, false, b1, (short)0, acc1, false, false);
        acc2 = __builtin_amdgcn_wmma_f32_16x16x32_bf16(false, a, false, b2, (short)0, acc2, false, false);
        acc3 = __builtin_amdgcn_wmma_f32_16x16x32_bf16(false, a, false, b3, (short)0, acc3, false, false);
    }
    // ---- h @ W_hh^T ------------------------------------------------------
    for (int k0 = 0; k0 < H_; k0 += 32) {
        v16bf a  = load_frag(h_in, arow,            H_, k0, lane);
        v16bf b0 = load_frag(W_hh, 0 * H_ + bcol,   H_, k0, lane);
        v16bf b1 = load_frag(W_hh, 1 * H_ + bcol,   H_, k0, lane);
        v16bf b2 = load_frag(W_hh, 2 * H_ + bcol,   H_, k0, lane);
        v16bf b3 = load_frag(W_hh, 3 * H_ + bcol,   H_, k0, lane);
        acc0 = __builtin_amdgcn_wmma_f32_16x16x32_bf16(false, a, false, b0, (short)0, acc0, false, false);
        acc1 = __builtin_amdgcn_wmma_f32_16x16x32_bf16(false, a, false, b1, (short)0, acc1, false, false);
        acc2 = __builtin_amdgcn_wmma_f32_16x16x32_bf16(false, a, false, b2, (short)0, acc2, false, false);
        acc3 = __builtin_amdgcn_wmma_f32_16x16x32_bf16(false, a, false, b3, (short)0, acc3, false, false);
    }

    // ---- fused gate activations (C/D layout: VGPR e -> row, lane -> col) --
    int n = bcol;
    float bi_ = bias[0 * H_ + n], bf_ = bias[1 * H_ + n];
    float bg_ = bias[2 * H_ + n], bo_ = bias[3 * H_ + n];
#pragma unroll
    for (int e = 0; e < 8; ++e) {
        int m = mt * 16 + e + ((lane >> 4) << 3);
        float iv = sigmoidf_(acc0[e] + bi_);
        float fv = sigmoidf_(acc1[e] + bf_);
        float gv = tanhf(acc2[e] + bg_);
        float ov = sigmoidf_(acc3[e] + bo_);
        size_t idx = (size_t)m * H_ + n;
        float c2 = fv * c_in[idx] + iv * gv;
        float hv = ov * tanhf(c2);
        c_out[idx]      = c2;
        h_out_f32[idx]  = hv;
        h_out_bf16[idx] = (__bf16)hv;
    }
}

// ---------------------------------------------------------------------------
// out[:, t, :] = h @ fc_W^T + fc_b   (M=128, N=H, K=H), written into d_out.
// ---------------------------------------------------------------------------
__global__ __launch_bounds__(128)
void fc_wmma(const __bf16* __restrict__ h,     // B x H (bf16)
             const __bf16* __restrict__ W,     // H x H (bf16)
             const float*  __restrict__ bias,  // H
             float* __restrict__ out,          // B x T x H  (d_out base)
             int t)
{
    int tid  = blockIdx.x * blockDim.x + threadIdx.x;
    int lane = tid & 31;
    int wid  = tid >> 5;
    int mt   = wid >> 6;
    int nt   = wid & 63;

    v8f acc = {};
    int arow = mt * 16 + (lane & 15);
    int brow = nt * 16 + (lane & 15);
    for (int k0 = 0; k0 < H_; k0 += 32) {
        v16bf a = load_frag(h, arow, H_, k0, lane);
        v16bf b = load_frag(W, brow, H_, k0, lane);
        acc = __builtin_amdgcn_wmma_f32_16x16x32_bf16(false, a, false, b, (short)0, acc, false, false);
    }
    int n = nt * 16 + (lane & 15);
    float bn = bias[n];
#pragma unroll
    for (int e = 0; e < 8; ++e) {
        int m = mt * 16 + e + ((lane >> 4) << 3);
        out[(size_t)m * T_ * H_ + (size_t)t * H_ + n] = acc[e] + bn;
    }
}

// ---------------------------------------------------------------------------
// Small helper kernels
// ---------------------------------------------------------------------------
__global__ void f32_to_bf16_k(const float* __restrict__ in, __bf16* __restrict__ out, int n) {
    int i = blockIdx.x * blockDim.x + threadIdx.x;
    if (i < n) out[i] = (__bf16)in[i];
}

__global__ void zero_f32_k(float* p, int n) {
    int i = blockIdx.x * blockDim.x + threadIdx.x;
    if (i < n) p[i] = 0.0f;
}

__global__ void zero_bf16_k(__bf16* p, int n) {
    int i = blockIdx.x * blockDim.x + threadIdx.x;
    if (i < n) p[i] = (__bf16)0.0f;
}

__global__ void copy_f32_k(const float* __restrict__ in, float* __restrict__ out, int n) {
    int i = blockIdx.x * blockDim.x + threadIdx.x;
    if (i < n) out[i] = in[i];
}

// x_all[t][b][:] = emb_bf16[inputs[b][t]]   (8-element bf16 vector chunks)
__global__ void embed_all_k(const int* __restrict__ inputs,
                            const __bf16* __restrict__ emb,
                            __bf16* __restrict__ x_all) {
    size_t ci = (size_t)blockIdx.x * blockDim.x + threadIdx.x;  // chunk id
    const int CPR = E_ / 8;                                     // 64 chunks/row
    int e8 = (int)(ci % CPR);
    int b  = (int)((ci / CPR) % B_);
    int t  = (int)(ci / ((size_t)CPR * B_));
    int tok = inputs[(size_t)b * T_ + t];
    const v8bf* src = reinterpret_cast<const v8bf*>(emb + (size_t)tok * E_ + e8 * 8);
    v8bf* dst = reinterpret_cast<v8bf*>(x_all + (size_t)t * B_ * E_ + (size_t)b * E_ + e8 * 8);
    *dst = *src;
}

// dh/dc = zeros with row (B-1) kept from encoder final state (in-place ok:
// each index touched by exactly one thread, read-then-write).
__global__ void make_dec_init_k(float* __restrict__ h_f,
                                __bf16* __restrict__ h_bf,
                                float* __restrict__ c_f) {
    int i = blockIdx.x * blockDim.x + threadIdx.x;     // over B_*H_
    bool last = (i >= (B_ - 1) * H_);
    float hv = last ? h_f[i] : 0.0f;
    float cv = last ? c_f[i] : 0.0f;
    h_f[i] = hv;
    h_bf[i] = (__bf16)hv;
    c_f[i] = cv;
}

// sym0 = inputs[:,0]; e_dec[b] = emb[sym0[b]]
__global__ void sym0_embed_k(const int* __restrict__ inputs,
                             int* __restrict__ sym,
                             const __bf16* __restrict__ emb,
                             __bf16* __restrict__ e_out) {
    int b = blockIdx.x;
    int s = inputs[(size_t)b * T_];
    if (threadIdx.x == 0) sym[b] = s;
    const __bf16* er = emb + (size_t)s * E_;
    __bf16* eo = e_out + (size_t)b * E_;
    for (int e = threadIdx.x; e < E_; e += blockDim.x) eo[e] = er[e];
}

// argmax over out[:, t, :] row (first-max tie-break, matching jnp.argmax),
// write symbol (as float) into d_out symbol section, and gather the embedding
// row for the next decoder step.
__global__ __launch_bounds__(256)
void argmax_embed_k(const float* __restrict__ dec_out, int t,
                    const __bf16* __restrict__ emb,
                    __bf16* __restrict__ e_out,
                    int* __restrict__ sym,
                    float* __restrict__ seq_out) {
    __shared__ float sval[256];
    __shared__ int   sidx[256];
    int b = blockIdx.x;
    const float* row = dec_out + (size_t)b * T_ * H_ + (size_t)t * H_;
    float bv = -__builtin_inff();
    int   bi = 0x7fffffff;
    for (int n = threadIdx.x; n < H_; n += 256) {
        float v = row[n];
        if (v > bv || (v == bv && n < bi)) { bv = v; bi = n; }
    }
    sval[threadIdx.x] = bv; sidx[threadIdx.x] = bi;
    __syncthreads();
    for (int s = 128; s > 0; s >>= 1) {
        if (threadIdx.x < s) {
            float v2 = sval[threadIdx.x + s];
            int   i2 = sidx[threadIdx.x + s];
            if (v2 > sval[threadIdx.x] ||
                (v2 == sval[threadIdx.x] && i2 < sidx[threadIdx.x])) {
                sval[threadIdx.x] = v2; sidx[threadIdx.x] = i2;
            }
        }
        __syncthreads();
    }
    int best = sidx[0];
    if (threadIdx.x == 0) {
        sym[b] = best;
        seq_out[(size_t)b * T_ + t] = (float)best;
    }
    const __bf16* er = emb + (size_t)best * E_;
    __bf16* eo = e_out + (size_t)b * E_;
    for (int e = threadIdx.x; e < E_; e += 256) eo[e] = er[e];
}

// ---------------------------------------------------------------------------
// Host orchestration
// ---------------------------------------------------------------------------
extern "C" void kernel_launch(void* const* d_in, const int* in_sizes, int n_in,
                              void* d_out, int out_size, void* d_ws, size_t ws_size,
                              hipStream_t stream) {
    (void)in_sizes; (void)n_in; (void)out_size; (void)ws_size;

    const int*   inputs    = (const int*)  d_in[0];
    const float* embedding = (const float*)d_in[1];
    const float* encWih    = (const float*)d_in[2];
    const float* encWhh    = (const float*)d_in[3];
    const float* encB      = (const float*)d_in[4];
    const float* decWih    = (const float*)d_in[5];
    const float* decWhh    = (const float*)d_in[6];
    const float* decB      = (const float*)d_in[7];
    const float* fcW       = (const float*)d_in[8];
    const float* fcB       = (const float*)d_in[9];
    float* out = (float*)d_out;

    const size_t OUT_HF  = (size_t)B_ * T_ * H_;
    const size_t OUT_CF  = OUT_HF + (size_t)B_ * H_;
    const size_t OUT_SYM = OUT_CF + (size_t)B_ * H_;

    // Bump allocator over workspace (256B-aligned slices).
    char* wsp = (char*)d_ws;
    size_t off = 0;
    auto alloc = [&](size_t bytes) -> void* {
        void* p = wsp + off;
        off += (bytes + 255) & ~(size_t)255;
        return p;
    };

    __bf16* emb_bf = (__bf16*)alloc((size_t)V_ * E_ * 2);
    __bf16* eWih   = (__bf16*)alloc((size_t)4 * H_ * E_ * 2);
    __bf16* eWhh   = (__bf16*)alloc((size_t)4 * H_ * H_ * 2);
    __bf16* dWih   = (__bf16*)alloc((size_t)4 * H_ * E_ * 2);
    __bf16* dWhh   = (__bf16*)alloc((size_t)4 * H_ * H_ * 2);
    __bf16* fcWb   = (__bf16*)alloc((size_t)H_ * H_ * 2);
    __bf16* x_all  = (__bf16*)alloc((size_t)T_ * B_ * E_ * 2);
    __bf16* h_bf[2] = { (__bf16*)alloc((size_t)B_ * H_ * 2),
                        (__bf16*)alloc((size_t)B_ * H_ * 2) };
    float*  h_f[2]  = { (float*)alloc((size_t)B_ * H_ * 4),
                        (float*)alloc((size_t)B_ * H_ * 4) };
    float*  c_f[2]  = { (float*)alloc((size_t)B_ * H_ * 4),
                        (float*)alloc((size_t)B_ * H_ * 4) };
    __bf16* e_dec   = (__bf16*)alloc((size_t)B_ * E_ * 2);
    int*    sym     = (int*)alloc((size_t)B_ * sizeof(int));

    // Weight / embedding conversion to bf16 (L2-resident working set ~27 MB).
    auto cvt = [&](const float* s, __bf16* d, int n) {
        f32_to_bf16_k<<<(n + 255) / 256, 256, 0, stream>>>(s, d, n);
    };
    cvt(embedding, emb_bf, V_ * E_);
    cvt(encWih, eWih, 4 * H_ * E_);
    cvt(encWhh, eWhh, 4 * H_ * H_);
    cvt(decWih, dWih, 4 * H_ * E_);
    cvt(decWhh, dWhh, 4 * H_ * H_);
    cvt(fcW, fcWb, H_ * H_);

    // Embed all encoder inputs: x_all[t][b][:] (bf16).
    embed_all_k<<<(T_ * B_ * (E_ / 8)) / 256, 256, 0, stream>>>(inputs, emb_bf, x_all);

    // h0 = 0, c0 = 0 (ping-pong buffer 0).
    zero_bf16_k<<<(B_ * H_ + 255) / 256, 256, 0, stream>>>(h_bf[0], B_ * H_);
    zero_f32_k <<<(B_ * H_ + 255) / 256, 256, 0, stream>>>(c_f[0], B_ * H_);

    // ---- Encoder scan (128 dependent steps, ping-pong state) -------------
    for (int t = 0; t < T_; ++t) {
        lstm_cell_wmma<<<128, 128, 0, stream>>>(
            x_all + (size_t)t * B_ * E_,
            h_bf[t & 1], c_f[t & 1],
            eWih, eWhh, encB,
            c_f[(t + 1) & 1], h_f[(t + 1) & 1], h_bf[(t + 1) & 1]);
    }
    // Final encoder state landed in buffer 0 (128 steps). Build dh/dc:
    // zeros everywhere except batch row B-1 (reference quirk).
    make_dec_init_k<<<(B_ * H_) / 256, 256, 0, stream>>>(h_f[0], h_bf[0], c_f[0]);

    // sym0 = inputs[:,0]; first decoder input embedding.
    sym0_embed_k<<<B_, 256, 0, stream>>>(inputs, sym, emb_bf, e_dec);

    // ---- Decoder scan with argmax feedback -------------------------------
    for (int t = 0; t < T_; ++t) {
        lstm_cell_wmma<<<128, 128, 0, stream>>>(
            e_dec,
            h_bf[t & 1], c_f[t & 1],
            dWih, dWhh, decB,
            c_f[(t + 1) & 1], h_f[(t + 1) & 1], h_bf[(t + 1) & 1]);
        fc_wmma<<<128, 128, 0, stream>>>(h_bf[(t + 1) & 1], fcWb, fcB, out, t);
        argmax_embed_k<<<B_, 256, 0, stream>>>(out, t, emb_bf, e_dec, sym,
                                               out + OUT_SYM);
    }

    // Final decoder state (after 128 steps -> buffer 0) to outputs.
    copy_f32_k<<<(B_ * H_) / 256, 256, 0, stream>>>(h_f[0], out + OUT_HF, B_ * H_);
    copy_f32_k<<<(B_ * H_) / 256, 256, 0, stream>>>(c_f[0], out + OUT_CF, B_ * H_);
}
